// PAKA2d_53944789238139
// MI455X (gfx1250) — matrix operand order repaired and checked by reading, not compiled, not run on hardware
//
#include <hip/hip_runtime.h>
#include <math.h>

typedef __attribute__((ext_vector_type(16))) __bf16 v16bf;
typedef __attribute__((ext_vector_type(8)))  float  v8f;

struct U32x8  { uint4 a, b; };

#define HDIM 64
#define WDIM 64
#define NCH  256
#define CM   64
#define KKTOT 2304
#define PSTR 2312   // patches row stride (ushorts): 2304 + 8 pad -> 4624 B (16B aligned)
#define YSTR 72     // yc/yd row stride (ushorts): 64 + 8 pad -> 144 B
#define XSTR 264    // center-x row stride (ushorts): 256 + 8 pad -> 528 B
#define CSTR 260    // c row stride (floats): 256 + 4 pad -> 1040 B

// LDS byte offsets (all 16B aligned)
#define LDS_PATCH 0
#define LDS_YC    73984
#define LDS_YD    76288
#define LDS_XC    78592
#define LDS_C     87040
#define LDS_D     103680
#define LDS_TOTAL 104704

// ws element offsets (bf16/ushort units), all byte offsets 16B-aligned
#define OFF_WC1 0        // 64 x 256
#define OFF_WC2 16384    // 256 x 64
#define OFF_WD1 32768    // 64 x 2304
#define OFF_WD2 180224   // 16 x 64 (rows 9..15 zero pad)
#define OFF_W   181248   // 256 x 2304

static __device__ __forceinline__ unsigned short f2bf(float f) {
  unsigned int u = __builtin_bit_cast(unsigned int, f);
  unsigned int r = u + 0x7FFFu + ((u >> 16) & 1u);   // round-to-nearest-even
  return (unsigned short)(r >> 16);
}
static __device__ __forceinline__ float bf2f(unsigned short s) {
  unsigned int u = ((unsigned int)s) << 16;
  return __builtin_bit_cast(float, u);
}

// 1 + tanh(z) == 2 * sigmoid(2z) == 2 * rcp(1 + exp2(-2*log2(e)*z))
// -> one v_exp_f32 + one v_rcp_f32 on the trans pipe (co-executes with WMMA)
static __device__ __forceinline__ float attn_fn(float z) {
  float e = __builtin_amdgcn_exp2f(-2.885390081777927f * z);
  return 2.0f * __builtin_amdgcn_rcpf(1.0f + e);
}

static __device__ __forceinline__ v8f wmma_bf16(v16bf a, v16bf b, v8f c) {
  // (neg_a, A, neg_b, B, c_mod, C, reuse_a, reuse_b)
  return __builtin_amdgcn_wmma_f32_16x16x32_bf16(false, a, false, b, (short)0, c,
                                                 false, false);
}

// A operand: bf16 row-major [M][K]; per ISA 16-bit A layout lane holds row
// m = mbase + (lane&15), K values {kh*8 .. kh*8+7} and {kh*8+16 .. kh*8+23}
static __device__ __forceinline__ v16bf load_A_global(
    const unsigned short* __restrict__ Wb, int rowStrideEls, int mbase, int kbase, int lane) {
  int m  = mbase + (lane & 15);
  int kh = lane >> 4;
  const unsigned short* row = Wb + (size_t)m * rowStrideEls + kbase + kh * 8;
  U32x8 t;
  t.a = *(const uint4*)(row);        // K: base+0..7
  t.b = *(const uint4*)(row + 16);   // K: base+16..23
  return __builtin_bit_cast(v16bf, t);
}

// B operand from LDS laid out [pixel][k] row-major: lane holds col p=lane&15,
// K = kbase + (lane>>4)*8 + {0..7} and {16..23} -> two ds_load_b128
static __device__ __forceinline__ v16bf load_B_lds(
    const unsigned short* base, int rowStrideEls, int kbase, int lane) {
  int p    = lane & 15;
  int koff = kbase + ((lane >> 4) << 3);
  const unsigned short* src = base + p * rowStrideEls + koff;
  U32x8 t;
  t.a = *(const uint4*)(src);
  t.b = *(const uint4*)(src + 16);
  return __builtin_bit_cast(v16bf, t);
}

__global__ void prep_weights_kernel(const float* __restrict__ w_c1,
                                    const float* __restrict__ w_c2,
                                    const float* __restrict__ w_d1,
                                    const float* __restrict__ w_d2,
                                    const float* __restrict__ Wm,
                                    unsigned short* __restrict__ ws) {
  int i = blockIdx.x * blockDim.x + threadIdx.x;
  if (i < 16384)  ws[OFF_WC1 + i] = f2bf(w_c1[i]);
  if (i < 16384)  ws[OFF_WC2 + i] = f2bf(w_c2[i]);
  if (i < 147456) ws[OFF_WD1 + i] = f2bf(w_d1[i]);
  if (i < 1024) {
    int r = i >> 6, c = i & 63;
    ws[OFF_WD2 + i] = (r < 9) ? f2bf(w_d2[r * 64 + c]) : (unsigned short)0;
  }
  if (i < 589824) ws[OFF_W + i] = f2bf(Wm[i]);
}

__global__ __launch_bounds__(256) void paka_fused_kernel(
    const float* __restrict__ x,
    const float* __restrict__ b_c1, const float* __restrict__ b_c2,
    const float* __restrict__ b_d1, const float* __restrict__ b_d2,
    const unsigned short* __restrict__ wsb,
    float* __restrict__ out) {
  extern __shared__ __align__(16) char smem[];
  unsigned short* patches = (unsigned short*)(smem + LDS_PATCH);
  unsigned short* ycL = (unsigned short*)(smem + LDS_YC);
  unsigned short* ydL = (unsigned short*)(smem + LDS_YD);
  unsigned short* xcL = (unsigned short*)(smem + LDS_XC);   // center taps [p][n]
  float* cL = (float*)(smem + LDS_C);
  float* dL = (float*)(smem + LDS_D);

  int bid  = blockIdx.x;                 // 2048 = 8 * 64 * 4
  int wt   = bid & 3;
  int hrow = (bid >> 2) & 63;
  int bb   = bid >> 8;
  int w0   = wt * 16;

  int t    = threadIdx.x;
  int lane = t & 31;
  int wave = t >> 5;
  int wv_s = __builtin_amdgcn_readfirstlane(wave);   // scalar branch, keep EXEC intact

  // ---- Phase A: build bf16 patch matrix [16 px][2304 = n*9+k] + center taps ----
  {
    int n = t;   // one channel per thread
    const float* xc = x + (size_t)(bb * NCH + n) * (HDIM * WDIM);
    float row[3][18];
#pragma unroll
    for (int dy = 0; dy < 3; ++dy) {
      int hh = hrow + dy - 1;
#pragma unroll
      for (int j = 0; j < 18; ++j) {
        int ww = w0 + j - 1;
        bool ok = (hh >= 0) && (hh < HDIM) && (ww >= 0) && (ww < WDIM);
        row[dy][j] = ok ? xc[hh * WDIM + ww] : 0.0f;
      }
    }
#pragma unroll
    for (int k = 0; k < 9; ++k) {
      int dy = k / 3, dx = k % 3;
#pragma unroll
      for (int p = 0; p < 16; ++p)
        patches[p * PSTR + n * 9 + k] = f2bf(row[dy][p + dx]);
    }
#pragma unroll
    for (int p = 0; p < 16; ++p)
      xcL[p * XSTR + n] = f2bf(row[1][p + 1]);   // center tap (dy=0, dx=0)
  }
  __syncthreads();

  // ---- Phase B: conv GEMM (waves 0-3) and channel GEMM1 (waves 4-7) ----
  if (wv_s < 4) {
    v8f acc = {};
    for (int kb = 0; kb < KKTOT; kb += 32) {
      v16bf a = load_A_global(wsb + OFF_WD1, KKTOT, wv_s * 16, kb, lane);
      v16bf b = load_B_lds(patches, PSTR, kb, lane);
      acc = wmma_bf16(a, b, acc);
    }
    int p = lane & 15, kh = lane >> 4;
#pragma unroll
    for (int r = 0; r < 8; ++r) {
      int m = wv_s * 16 + r + kh * 8;
      float v = acc[r] + b_d1[m];
      ydL[p * YSTR + m] = f2bf(v > 0.f ? v : 0.f);
    }
  } else {
    int mt = wv_s - 4;
    v8f acc = {};
#pragma unroll
    for (int kb = 0; kb < NCH; kb += 32) {
      v16bf a = load_A_global(wsb + OFF_WC1, NCH, mt * 16, kb, lane);
      v16bf b = load_B_lds(xcL, XSTR, kb, lane);
      acc = wmma_bf16(a, b, acc);
    }
    int p = lane & 15, kh = lane >> 4;
#pragma unroll
    for (int r = 0; r < 8; ++r) {
      int m = mt * 16 + r + kh * 8;
      float v = acc[r] + b_c1[m];
      ycL[p * YSTR + m] = f2bf(v > 0.f ? v : 0.f);
    }
  }
  __syncthreads();

  // ---- Phase C: GEMM2 (c = w_c2 @ yc), dual accumulators; d = w_d2 @ yd on wave 0 ----
  {
    int mt0 = wave * 2;
    v8f acc0 = {}, acc1 = {};
#pragma unroll
    for (int kb = 0; kb < CM; kb += 32) {
      v16bf b  = load_B_lds(ycL, YSTR, kb, lane);
      v16bf a0 = load_A_global(wsb + OFF_WC2, CM, mt0 * 16, kb, lane);
      v16bf a1 = load_A_global(wsb + OFF_WC2, CM, mt0 * 16 + 16, kb, lane);
      acc0 = wmma_bf16(a0, b, acc0);
      acc1 = wmma_bf16(a1, b, acc1);
    }
    int p = lane & 15, kh = lane >> 4;
#pragma unroll
    for (int r = 0; r < 8; ++r) {
      int m0 = mt0 * 16 + r + kh * 8;
      cL[p * CSTR + m0]      = acc0[r] + b_c2[m0];
      cL[p * CSTR + m0 + 16] = acc1[r] + b_c2[m0 + 16];
    }
    if (wv_s == 0) {
      v8f acc = {};
#pragma unroll
      for (int kb = 0; kb < CM; kb += 32) {
        v16bf a = load_A_global(wsb + OFF_WD2, CM, 0, kb, lane);
        v16bf b = load_B_lds(ydL, YSTR, kb, lane);
        acc = wmma_bf16(a, b, acc);
      }
#pragma unroll
      for (int r = 0; r < 8; ++r) {
        int m = r + kh * 8;
        if (m < 9) dL[p * 16 + m] = acc[r] + b_d2[m];
      }
    }
  }
  __syncthreads();

  // ---- Phase D: modulate patches in place: *= 1 + tanh(d[k] + c[n]) ----
  {
    int n = t;
#pragma unroll
    for (int p = 0; p < 16; ++p) {
      float cn = cL[p * CSTR + n];
#pragma unroll
      for (int k = 0; k < 9; ++k) {
        float attn = attn_fn(dL[p * 16 + k] + cn);
        unsigned short* el = &patches[p * PSTR + n * 9 + k];
        *el = f2bf(bf2f(*el) * attn);
      }
    }
  }
  __syncthreads();

  // ---- Phase E: main GEMM out(256 x 16) = W(256 x 2304) @ modulated patches ----
  // Dual accumulators: one LDS B load feeds two W tiles (two wmma per chunk).
  {
    int mt0 = wave * 2;
    v8f acc0 = {}, acc1 = {};
    for (int kb = 0; kb < KKTOT; kb += 32) {
      v16bf b  = load_B_lds(patches, PSTR, kb, lane);
      v16bf a0 = load_A_global(wsb + OFF_W, KKTOT, mt0 * 16, kb, lane);
      v16bf a1 = load_A_global(wsb + OFF_W, KKTOT, mt0 * 16 + 16, kb, lane);
      acc0 = wmma_bf16(a0, b, acc0);
      acc1 = wmma_bf16(a1, b, acc1);
    }
    int p = lane & 15, kh = lane >> 4;
    size_t obase = (((size_t)bb * 256 + mt0 * 16) * HDIM + hrow) * WDIM + w0 + p;
#pragma unroll
    for (int r = 0; r < 8; ++r) {
      int m = r + kh * 8;
      out[obase + (size_t)m * (HDIM * WDIM)]        = acc0[r];
      out[obase + (size_t)(m + 16) * (HDIM * WDIM)] = acc1[r];
    }
  }
}

extern "C" void kernel_launch(void* const* d_in, const int* in_sizes, int n_in,
                              void* d_out, int out_size, void* d_ws, size_t ws_size,
                              hipStream_t stream) {
  (void)in_sizes; (void)n_in; (void)out_size; (void)ws_size;
  const float* x    = (const float*)d_in[0];
  const float* w_c1 = (const float*)d_in[1];
  const float* b_c1 = (const float*)d_in[2];
  const float* w_c2 = (const float*)d_in[3];
  const float* b_c2 = (const float*)d_in[4];
  const float* w_d1 = (const float*)d_in[5];
  const float* b_d1 = (const float*)d_in[6];
  const float* w_d2 = (const float*)d_in[7];
  const float* b_d2 = (const float*)d_in[8];
  const float* Wm   = (const float*)d_in[9];
  float* out = (float*)d_out;
  unsigned short* wsb = (unsigned short*)d_ws;

  // Convert all weights to bf16 (W/w_d1 already in n*9+k row-major order).
  prep_weights_kernel<<<2304, 256, 0, stream>>>(w_c1, w_c2, w_d1, w_d2, Wm, wsb);

  // One workgroup per (b, h, 16-wide w strip): 8*64*4 = 2048 blocks.
  paka_fused_kernel<<<2048, 256, LDS_TOTAL, stream>>>(x, b_c1, b_c2, b_d1, b_d2,
                                                      wsb, out);
}
